// _GQAttention_60679297958612
// MI455X (gfx1250) — matrix operand order repaired and checked by reading, not compiled, not run on hardware
//
#include <hip/hip_runtime.h>
#include <hip/hip_fp16.h>

// ---------------- problem constants ----------------
#define B_   4
#define T_   2048
#define D_   1024
#define HQ_  16
#define HKV_ 4
#define HD_  64
#define NKV_ (2 * HKV_ * HD_)   // 512
#define NTOT (D_ + NKV_)        // 1536
#define M_   (B_ * T_)          // 8192

typedef __attribute__((ext_vector_type(16))) _Float16 v16h;
typedef __attribute__((ext_vector_type(8)))  _Float16 v8h;
typedef __attribute__((ext_vector_type(8)))  float    v8f;

union Frag16 { v16h v; v8h h[2]; };

__device__ __forceinline__ v8f wmma_f16(v16h a, v16h b, v8f c) {
  // D = A(16x32) * B(32x16) + C(16x16), f32 accumulate
  return __builtin_amdgcn_wmma_f32_16x16x32_f16(
      /*neg_a=*/false, a, /*neg_b=*/false, b,
      /*c_mod=*/(short)0, c, /*reuse_a=*/false, /*reuse_b=*/false);
}

// ---- CDNA5 async global->LDS copy (GVS mode: SGPR base + VGPR offset) ----
// Copies 16 bytes per lane directly into LDS, tracked by ASYNCcnt.
// No "memory" clobber here: ordering vs consumers is enforced by
// s_wait_asynccnt + the workgroup barrier (volatile asm is not reordered
// across either).
__device__ __forceinline__ void async_ld_b128(void* lds_ptr, unsigned goff,
                                              const void* sbase) {
  unsigned lds_addr = (unsigned)(uintptr_t)lds_ptr;  // low word = LDS offset
  asm volatile("global_load_async_to_lds_b128 %0, %1, %2"
               :
               : "v"(lds_addr), "v"(goff), "s"(sbase));
}
__device__ __forceinline__ void wait_async0() {
#if __has_builtin(__builtin_amdgcn_s_wait_asynccnt)
  __builtin_amdgcn_s_wait_asynccnt(0);
#else
  asm volatile("s_wait_asynccnt 0x0" ::: "memory");
#endif
}

__device__ __forceinline__ float half_reduce_max(float x) {
  x = fmaxf(x, __shfl_xor(x, 1));
  x = fmaxf(x, __shfl_xor(x, 2));
  x = fmaxf(x, __shfl_xor(x, 4));
  x = fmaxf(x, __shfl_xor(x, 8));
  return x;   // reduced across each 16-lane half independently
}
__device__ __forceinline__ float half_reduce_sum(float x) {
  x += __shfl_xor(x, 1);
  x += __shfl_xor(x, 2);
  x += __shfl_xor(x, 4);
  x += __shfl_xor(x, 8);
  return x;
}

// =====================================================================
// Kernel 1: fused QKV projection + RoPE (double-buffered pipeline).
//   C[m, n] = dot(x[m, :], W[n, :])   m in [0,8192), n in [0,1536)
//   n <  1024 -> Q feature  (RoPE + 1/8 scale), layout [b][hq][t][d] f16
//   n >= 1024 -> KV feature (K gets RoPE),      layout [b][hkv][t][d] f16
// Block: 256 threads (8 waves), tile 128(M) x 64(N), K-step 32.
// =====================================================================
__global__ __launch_bounds__(256)
void qkv_rope_kernel(const float* __restrict__ x,
                     const float* __restrict__ Wq,
                     const float* __restrict__ Wkv,
                     const float* __restrict__ cosp,
                     const float* __restrict__ sinp,
                     _Float16* __restrict__ Qh,
                     _Float16* __restrict__ Kh,
                     _Float16* __restrict__ Vh) {
  __shared__ __attribute__((aligned(16))) _Float16 As[2][128][40];
  __shared__ __attribute__((aligned(16))) _Float16 Bs[2][32][72];
  __shared__ __attribute__((aligned(16))) float    Cs[8][32][33];

  const int tid  = threadIdx.x;
  const int wave = tid >> 5;
  const int lane = tid & 31;
  const int lo   = lane & 15;
  const int hi   = lane >> 4;
  const int wm   = wave & 3;     // 4 M sub-blocks of 32
  const int wn   = wave >> 2;    // 2 N sub-blocks of 32

  const int rb = blockIdx.x / (NTOT / 64);
  const int cb = blockIdx.x % (NTOT / 64);
  const int m0 = rb * 128;
  const int n0 = cb * 64;

  // per-thread staging coordinates
  int arow[4], ac4[4];
#pragma unroll
  for (int it = 0; it < 4; ++it) {
    int slot = tid + 256 * it;     // 0..1023
    arow[it] = slot >> 3;
    ac4[it]  = (slot & 7) * 4;
  }
  int bnr[2], bc4[2];
  const float* brow[2];
#pragma unroll
  for (int it = 0; it < 2; ++it) {
    int slot = tid + 256 * it;     // 0..511
    bnr[it]  = slot >> 3;
    bc4[it]  = (slot & 7) * 4;
    int n    = n0 + bnr[it];
    brow[it] = (n < D_) ? (Wq + (size_t)n * D_) : (Wkv + (size_t)(n - D_) * D_);
  }

  float4 areg[4], breg[2];
  auto loadAB = [&](int k0) {
#pragma unroll
    for (int it = 0; it < 4; ++it)
      areg[it] = *(const float4*)(x + (size_t)(m0 + arow[it]) * D_ + k0 + ac4[it]);
#pragma unroll
    for (int it = 0; it < 2; ++it)
      breg[it] = *(const float4*)(brow[it] + k0 + bc4[it]);
  };
  auto storeAB = [&](int buf) {
#pragma unroll
    for (int it = 0; it < 4; ++it) {
      As[buf][arow[it]][ac4[it] + 0] = (_Float16)areg[it].x;
      As[buf][arow[it]][ac4[it] + 1] = (_Float16)areg[it].y;
      As[buf][arow[it]][ac4[it] + 2] = (_Float16)areg[it].z;
      As[buf][arow[it]][ac4[it] + 3] = (_Float16)areg[it].w;
    }
#pragma unroll
    for (int it = 0; it < 2; ++it) {
      Bs[buf][bc4[it] + 0][bnr[it]] = (_Float16)breg[it].x;
      Bs[buf][bc4[it] + 1][bnr[it]] = (_Float16)breg[it].y;
      Bs[buf][bc4[it] + 2][bnr[it]] = (_Float16)breg[it].z;
      Bs[buf][bc4[it] + 3][bnr[it]] = (_Float16)breg[it].w;
    }
  };

  v8f acc[2][2];
#pragma unroll
  for (int i = 0; i < 2; ++i)
#pragma unroll
    for (int j = 0; j < 2; ++j) acc[i][j] = (v8f){};

  // prologue: stage k-tile 0
  loadAB(0);
  storeAB(0);
  __syncthreads();

  const int KT = D_ / 32;
  for (int kt = 0; kt < KT; ++kt) {
    const int cur = kt & 1, nxt = cur ^ 1;
    if (kt + 1 < KT) loadAB((kt + 1) * 32);   // global loads overlap WMMAs

#pragma unroll
    for (int ti = 0; ti < 2; ++ti) {
      Frag16 a;
      int row = 32 * wm + 16 * ti + lo;
      a.h[0] = *(const v8h*)&As[cur][row][8 * hi];
      a.h[1] = *(const v8h*)&As[cur][row][16 + 8 * hi];
#pragma unroll
      for (int tj = 0; tj < 2; ++tj) {
        Frag16 b;
#pragma unroll
        for (int s = 0; s < 16; ++s)
          b.v[s] = Bs[cur][(s < 8) ? (8 * hi + s) : (16 + 8 * hi + s - 8)][32 * wn + 16 * tj + lo];
        acc[ti][tj] = wmma_f16(a.v, b.v, acc[ti][tj]);
      }
    }

    if (kt + 1 < KT) storeAB(nxt);
    __syncthreads();
  }

  // ---- epilogue: C tile -> per-wave LDS patch, then RoPE + scatter ----
#pragma unroll
  for (int ti = 0; ti < 2; ++ti)
#pragma unroll
    for (int tj = 0; tj < 2; ++tj)
#pragma unroll
      for (int r = 0; r < 8; ++r)
        Cs[wave][16 * ti + r + 8 * hi][16 * tj + lo] = acc[ti][tj][r];

  const int m_base = m0 + 32 * wm;
  const int n_base = n0 + 32 * wn;
  for (int idx = lane; idx < 32 * 16; idx += 32) {
    int rrow = idx >> 4;          // 0..31
    int cc   = (idx & 15) * 2;    // even column (RoPE pair base)
    float e = Cs[wave][rrow][cc];
    float o = Cs[wave][rrow][cc + 1];
    int m = m_base + rrow;
    int b = m >> 11;              // / T_
    int t = m & (T_ - 1);
    int n = n_base + cc;
    if (n < D_) {                 // Q feature: RoPE + fold 1/sqrt(HD)
      int h = n >> 6, d = n & 63;
      float c0 = cosp[t * HD_ + d],     s0 = sinp[t * HD_ + d];
      float c1 = cosp[t * HD_ + d + 1], s1 = sinp[t * HD_ + d + 1];
      float q0 = (e * c0 - o * s0) * 0.125f;
      float q1 = (o * c1 + e * s1) * 0.125f;
      _Float16* dst = Qh + (((size_t)b * HQ_ + h) * T_ + t) * HD_ + d;
      dst[0] = (_Float16)q0;
      dst[1] = (_Float16)q1;
    } else {
      int n2 = n - D_;
      int g = n2 >> 7;            // kv head
      int rem = n2 & 127;
      int d = rem & 63;
      if ((rem >> 6) == 0) {      // K: RoPE
        float c0 = cosp[t * HD_ + d],     s0 = sinp[t * HD_ + d];
        float c1 = cosp[t * HD_ + d + 1], s1 = sinp[t * HD_ + d + 1];
        float k0v = e * c0 - o * s0;
        float k1v = o * c1 + e * s1;
        _Float16* dst = Kh + (((size_t)b * HKV_ + g) * T_ + t) * HD_ + d;
        dst[0] = (_Float16)k0v;
        dst[1] = (_Float16)k1v;
      } else {                    // V: plain convert
        _Float16* dst = Vh + (((size_t)b * HKV_ + g) * T_ + t) * HD_ + d;
        dst[0] = (_Float16)e;
        dst[1] = (_Float16)o;
      }
    }
  }
}

// =====================================================================
// Kernel 2: flash-style GQA attention, double-buffered.
// grid = B*HQ*(T/128); block = 256 (8 waves). Wave owns 16 q rows.
// Next key-tile's V goes via CDNA5 async global->LDS DMA and K loads are
// issued before the current tile's WMMAs; transpose stores land after.
// =====================================================================
__global__ __launch_bounds__(256)
void attn_kernel(const _Float16* __restrict__ Qh,
                 const _Float16* __restrict__ Kh,
                 const _Float16* __restrict__ Vh,
                 _Float16* __restrict__ Oh) {
  __shared__ __attribute__((aligned(16))) _Float16 Kts[2][64][72];   // [d][key]
  __shared__ __attribute__((aligned(16))) _Float16 Vs[2][64][72];    // [key][d]
  __shared__ __attribute__((aligned(16))) _Float16 Ps[8][16][72];    // per-wave

  const int tid  = threadIdx.x;
  const int wave = tid >> 5;
  const int lane = tid & 31;
  const int lo   = lane & 15;
  const int hi   = lane >> 4;

  const int qblk = blockIdx.x & 15;
  const int bh   = blockIdx.x >> 4;
  const int h    = bh & (HQ_ - 1);
  const int b    = bh >> 4;
  const int g    = h >> 2;                   // kv head (r = HQ/HKV = 4)
  const int q0   = qblk * 128 + wave * 16;   // first q row of this wave

  const _Float16* kbase = Kh + ((size_t)b * HKV_ + g) * T_ * HD_;
  const _Float16* vbase = Vh + ((size_t)b * HKV_ + g) * T_ * HD_;

  // per-thread staging coordinates (512 chunks of 16B per tile)
  int keyA[2], c8A[2];
#pragma unroll
  for (int it = 0; it < 2; ++it) {
    int slot = tid + 256 * it;
    keyA[it] = slot >> 3;          // 0..63
    c8A[it]  = (slot & 7) * 8;     // d offset
  }

  v8h kreg[2];
  auto issueV = [&](int t0, int buf) {
    const _Float16* vtile = vbase + (size_t)t0 * HD_;  // uniform (SGPR) base
#pragma unroll
    for (int it = 0; it < 2; ++it)
      async_ld_b128(&Vs[buf][keyA[it]][c8A[it]],
                    (unsigned)((keyA[it] * HD_ + c8A[it]) * 2), vtile);
  };
  auto loadK = [&](int t0) {
#pragma unroll
    for (int it = 0; it < 2; ++it)
      kreg[it] = *(const v8h*)(kbase + (size_t)(t0 + keyA[it]) * HD_ + c8A[it]);
  };
  auto storeK = [&](int buf) {
#pragma unroll
    for (int it = 0; it < 2; ++it)
#pragma unroll
      for (int j = 0; j < 8; ++j)
        Kts[buf][c8A[it] + j][keyA[it]] = kreg[it][j];
  };

  // ---- load Q fragments (1/8 scale already folded in) ----
  Frag16 qf[2];
  {
    const _Float16* qrow =
        Qh + (((size_t)b * HQ_ + h) * T_ + q0 + lo) * HD_;
#pragma unroll
    for (int c = 0; c < 2; ++c) {
      qf[c].h[0] = *(const v8h*)(qrow + 32 * c + 8 * hi);
      qf[c].h[1] = *(const v8h*)(qrow + 32 * c + 16 + 8 * hi);
    }
  }

  v8f o[4];
#pragma unroll
  for (int j = 0; j < 4; ++j) o[j] = (v8f){};
  float mrow[8], lrow[8];
#pragma unroll
  for (int r = 0; r < 8; ++r) { mrow[r] = -1e30f; lrow[r] = 0.f; }

  // prologue: stage key-tile 0
  issueV(0, 0);
  loadK(0);
  storeK(0);
  wait_async0();
  __syncthreads();

  const int NT = T_ / 64;
  for (int kt = 0; kt < NT; ++kt) {
    const int cur = kt & 1, nxt = cur ^ 1;
    if (kt + 1 < NT) {             // prefetch next tile (DMA + K loads)
      issueV((kt + 1) * 64, nxt);
      loadK((kt + 1) * 64);
    }

    // ---- S = Q * K^T : 4 n-tiles x 2 k-chunks ----
    v8f s[4];
#pragma unroll
    for (int n = 0; n < 4; ++n) {
      s[n] = (v8f){};
#pragma unroll
      for (int c = 0; c < 2; ++c) {
        Frag16 bk;
#pragma unroll
        for (int s = 0; s < 16; ++s)
          bk.v[s] = Kts[cur][32 * c + ((s < 8) ? (8 * hi + s) : (16 + 8 * hi + s - 8))][16 * n + lo];
        s[n] = wmma_f16(qf[c].v, bk.v, s[n]);
      }
    }

    // ---- online softmax ----
    float mnew[8], alpha[8], rs[8];
#pragma unroll
    for (int r = 0; r < 8; ++r) {
      float cm = fmaxf(fmaxf(s[0][r], s[1][r]), fmaxf(s[2][r], s[3][r]));
      cm = half_reduce_max(cm);
      mnew[r]  = fmaxf(mrow[r], cm);
      alpha[r] = __expf(mrow[r] - mnew[r]);
      rs[r] = 0.f;
    }
#pragma unroll
    for (int n = 0; n < 4; ++n)
#pragma unroll
      for (int r = 0; r < 8; ++r) {
        float p = __expf(s[n][r] - mnew[r]);
        rs[r] += p;
        Ps[wave][r + 8 * hi][16 * n + lo] = (_Float16)p;
      }
#pragma unroll
    for (int r = 0; r < 8; ++r) {
      lrow[r] = alpha[r] * lrow[r] + half_reduce_sum(rs[r]);
      mrow[r] = mnew[r];
    }
#pragma unroll
    for (int j = 0; j < 4; ++j)
#pragma unroll
      for (int r = 0; r < 8; ++r) o[j][r] *= alpha[r];

    // ---- O += P * V ----
#pragma unroll
    for (int c = 0; c < 2; ++c) {
      Frag16 pf;
      pf.h[0] = *(const v8h*)&Ps[wave][lo][32 * c + 8 * hi];
      pf.h[1] = *(const v8h*)&Ps[wave][lo][32 * c + 16 + 8 * hi];
#pragma unroll
      for (int j = 0; j < 4; ++j) {
        Frag16 bv;
#pragma unroll
        for (int s = 0; s < 16; ++s)
          bv.v[s] = Vs[cur][32 * c + ((s < 8) ? (8 * hi + s) : (16 + 8 * hi + s - 8))][16 * j + lo];
        o[j] = wmma_f16(pf.v, bv.v, o[j]);
      }
    }

    if (kt + 1 < NT) storeK(nxt);  // next-tile transpose stores after compute
    wait_async0();
    __syncthreads();
  }

  // ---- normalize + store O in [b][t][h][d] (row-major [B*T][D]) f16 ----
  float inv[8];
#pragma unroll
  for (int r = 0; r < 8; ++r) inv[r] = 1.0f / lrow[r];
#pragma unroll
  for (int j = 0; j < 4; ++j)
#pragma unroll
    for (int r = 0; r < 8; ++r) {
      int q = q0 + r + 8 * hi;
      Oh[(size_t)(b * T_ + q) * D_ + h * HD_ + 16 * j + lo] =
          (_Float16)(o[j][r] * inv[r]);
    }
}

// =====================================================================
// Kernel 3: output projection  out = O @ Wo^T  (f16 x f16 -> f32),
// double-buffered; A tile staged via async global->LDS DMA.
// Block tile 128x64, K-step 32. grid = 64 * 16.
// =====================================================================
__global__ __launch_bounds__(256)
void oproj_kernel(const _Float16* __restrict__ Oh,
                  const float* __restrict__ Wo,
                  float* __restrict__ out) {
  __shared__ __attribute__((aligned(16))) _Float16 As[2][128][40];
  __shared__ __attribute__((aligned(16))) _Float16 Bs[2][32][72];

  const int tid  = threadIdx.x;
  const int wave = tid >> 5;
  const int lane = tid & 31;
  const int lo   = lane & 15;
  const int hi   = lane >> 4;
  const int wm   = wave & 3;
  const int wn   = wave >> 2;

  const int rb = blockIdx.x >> 4;
  const int cb = blockIdx.x & 15;
  const int m0 = rb * 128;
  const int n0 = cb * 64;

  int arow[2], ac8[2], bnr[2], bc4[2];
#pragma unroll
  for (int it = 0; it < 2; ++it) {
    int chunk = tid + 256 * it;    // 0..511
    arow[it] = chunk >> 2;
    ac8[it]  = (chunk & 3) * 8;
    bnr[it]  = chunk >> 3;
    bc4[it]  = (chunk & 7) * 4;
  }

  float4 breg[2];
  auto issueA = [&](int k0, int buf) {
    const _Float16* abase = Oh + (size_t)m0 * D_ + k0;  // uniform base
#pragma unroll
    for (int it = 0; it < 2; ++it)
      async_ld_b128(&As[buf][arow[it]][ac8[it]],
                    (unsigned)((arow[it] * D_ + ac8[it]) * 2), abase);
  };
  auto loadB = [&](int k0) {
#pragma unroll
    for (int it = 0; it < 2; ++it)
      breg[it] = *(const float4*)(Wo + (size_t)(n0 + bnr[it]) * D_ + k0 + bc4[it]);
  };
  auto storeB = [&](int buf) {
#pragma unroll
    for (int it = 0; it < 2; ++it) {
      Bs[buf][bc4[it] + 0][bnr[it]] = (_Float16)breg[it].x;
      Bs[buf][bc4[it] + 1][bnr[it]] = (_Float16)breg[it].y;
      Bs[buf][bc4[it] + 2][bnr[it]] = (_Float16)breg[it].z;
      Bs[buf][bc4[it] + 3][bnr[it]] = (_Float16)breg[it].w;
    }
  };

  v8f acc[2][2];
#pragma unroll
  for (int i = 0; i < 2; ++i)
#pragma unroll
    for (int j = 0; j < 2; ++j) acc[i][j] = (v8f){};

  // prologue
  issueA(0, 0);
  loadB(0);
  storeB(0);
  wait_async0();
  __syncthreads();

  const int KT = D_ / 32;
  for (int kt = 0; kt < KT; ++kt) {
    const int cur = kt & 1, nxt = cur ^ 1;
    if (kt + 1 < KT) {
      issueA((kt + 1) * 32, nxt);   // DMA overlaps WMMAs
      loadB((kt + 1) * 32);
    }

#pragma unroll
    for (int ti = 0; ti < 2; ++ti) {
      Frag16 a;
      int row = 32 * wm + 16 * ti + lo;
      a.h[0] = *(const v8h*)&As[cur][row][8 * hi];
      a.h[1] = *(const v8h*)&As[cur][row][16 + 8 * hi];
#pragma unroll
      for (int tj = 0; tj < 2; ++tj) {
        Frag16 bfr;
#pragma unroll
        for (int s = 0; s < 16; ++s)
          bfr.v[s] = Bs[cur][(s < 8) ? (8 * hi + s) : (16 + 8 * hi + s - 8)][32 * wn + 16 * tj + lo];
        acc[ti][tj] = wmma_f16(a.v, bfr.v, acc[ti][tj]);
      }
    }

    if (kt + 1 < KT) storeB(nxt);
    wait_async0();
    __syncthreads();
  }

#pragma unroll
  for (int ti = 0; ti < 2; ++ti)
#pragma unroll
    for (int tj = 0; tj < 2; ++tj)
#pragma unroll
      for (int r = 0; r < 8; ++r) {
        int m = m0 + 32 * wm + 16 * ti + r + 8 * hi;
        int n = n0 + 32 * wn + 16 * tj + lo;
        out[(size_t)m * D_ + n] = acc[ti][tj][r];
      }
}

// =====================================================================
extern "C" void kernel_launch(void* const* d_in, const int* in_sizes, int n_in,
                              void* d_out, int out_size, void* d_ws, size_t ws_size,
                              hipStream_t stream) {
  const float* x    = (const float*)d_in[0];
  const float* cosp = (const float*)d_in[1];
  const float* sinp = (const float*)d_in[2];
  const float* Wq   = (const float*)d_in[3];
  const float* Wkv  = (const float*)d_in[4];
  const float* Wo   = (const float*)d_in[5];

  _Float16* Qh = (_Float16*)d_ws;                               // 16 MB
  _Float16* Kh = Qh + (size_t)B_ * HQ_ * T_ * HD_;              // 4 MB
  _Float16* Vh = Kh + (size_t)B_ * HKV_ * T_ * HD_;             // 4 MB
  _Float16* Oh = Vh + (size_t)B_ * HKV_ * T_ * HD_;             // 16 MB

  qkv_rope_kernel<<<(M_ / 128) * (NTOT / 64), 256, 0, stream>>>(
      x, Wq, Wkv, cosp, sinp, Qh, Kh, Vh);
  attn_kernel<<<B_ * HQ_ * (T_ / 128), 256, 0, stream>>>(Qh, Kh, Vh, Oh);
  oproj_kernel<<<(M_ / 128) * (D_ / 64), 256, 0, stream>>>(Oh, Wo, (float*)d_out);
}
